// Generator_3040836845603
// MI455X (gfx1250) — matrix operand order repaired and checked by reading, not compiled, not run on hardware
//
#include <hip/hip_runtime.h>

// ---------------------------------------------------------------------------
// Point3DConv fused kernel for gfx1250 (MI455X), wave32 WMMA.
// out[b,o,n] = sum_k relu(BNp(Wp @ (relu(BNd(Wd @ delta)) * relu(BNf(Wf @ feat_gather)))))
// ---------------------------------------------------------------------------

typedef _Float16 v16h __attribute__((ext_vector_type(16)));
typedef _Float16 v8h  __attribute__((ext_vector_type(8)));
typedef __fp16   v2hf __attribute__((ext_vector_type(2)));   // cvt_pkrtz result type
typedef __fp16   v4hf __attribute__((ext_vector_type(4)));
typedef __fp16   v8hf __attribute__((ext_vector_type(8)));
typedef float    v8f  __attribute__((ext_vector_type(8)));
typedef float    v2f  __attribute__((ext_vector_type(2)));

#define BB   2
#define NN   20000
#define KK   17
#define CINV 128
#define GV   128
#define EPSV 1e-5f

// LDS layout (bytes). Row stride of 136 halves (272 B) breaks bank conflicts.
#define WSTRIDE  136
#define OFF_WF   0                    // 128*136 halves = 34816 B
#define OFF_WP   34816                // 34816 B
#define OFF_WD   69632                // 128 * float4 = 2048 B
#define OFF_CF   71680                // 128 floats
#define OFF_CP   72192                // 128 floats
#define OFF_AF   72704                // 128 floats (staging only)
#define OFF_AP   73216                // 128 floats (staging only)
#define OFF_WAVE 73728                // per-wave fbuf[4] then pbuf[4]
#define FB_BYTES 4352                 // 16 cols * 136 halves * 2 B
#define SMEM_TOTAL (OFF_WAVE + 8 * FB_BYTES)   // 108544 B

// Per-wave LDS ordering fence: CDNA5 DS ops from one wave complete in order,
// so same-wave store->load through LDS only needs compiler ordering + dscnt.
#define WAVE_LDS_FENCE() asm volatile("s_wait_dscnt 0" ::: "memory")

// Build a v16h WMMA operand from two 16-byte LDS chunks 'gap' halves apart.
__device__ __forceinline__ v16h ld_op(const _Float16* p, int gap) {
  v8h lo = *(const v8h*)(p);
  v8h hi = *(const v8h*)(p + gap);
  return __builtin_shufflevector(lo, hi, 0,1,2,3,4,5,6,7,8,9,10,11,12,13,14,15);
}

// Pack 4 cvt_pkrtz pairs into one 16-byte vector.
__device__ __forceinline__ v8hf pack8(v2hf p0, v2hf p1, v2hf p2, v2hf p3) {
  v4hf lo = __builtin_shufflevector(p0, p1, 0, 1, 2, 3);
  v4hf hi = __builtin_shufflevector(p2, p3, 0, 1, 2, 3);
  return __builtin_shufflevector(lo, hi, 0, 1, 2, 3, 4, 5, 6, 7);
}

__global__ void __launch_bounds__(128, 1)
p3dconv_wmma(const float* __restrict__ feats, const float* __restrict__ pts,
             const int* __restrict__ knn,
             const float* __restrict__ Wd, const float* __restrict__ bd,
             const float* __restrict__ gd, const float* __restrict__ betad,
             const float* __restrict__ md, const float* __restrict__ vd,
             const float* __restrict__ Wf, const float* __restrict__ bf,
             const float* __restrict__ gf, const float* __restrict__ betaf,
             const float* __restrict__ mf, const float* __restrict__ vf,
             const float* __restrict__ Wp, const float* __restrict__ bp,
             const float* __restrict__ gp, const float* __restrict__ betap,
             const float* __restrict__ mp, const float* __restrict__ vp,
             float* __restrict__ out)
{
  extern __shared__ char smem[];
  _Float16* wf16 = (_Float16*)(smem + OFF_WF);
  _Float16* wp16 = (_Float16*)(smem + OFF_WP);
  float*    wd4  = (float*)(smem + OFF_WD);
  float*    cfv  = (float*)(smem + OFF_CF);
  float*    cpv  = (float*)(smem + OFF_CP);
  float*    afv  = (float*)(smem + OFF_AF);
  float*    apv  = (float*)(smem + OFF_AP);

  const int tid = threadIdx.x;

  // ---- Stage folded BN params: y = (Wx+b)*a + (beta - m*a) => scale a into W,
  //      keep c = b*a + beta - m*a as additive term (delta-path c goes in Wd col 3).
  if (tid < 128) {
    const int o = tid;
    const float ad = gd[o] * rsqrtf(vd[o] + EPSV);
    const float cd = bd[o] * ad + betad[o] - md[o] * ad;
    const float af = gf[o] * rsqrtf(vf[o] + EPSV);
    const float ap = gp[o] * rsqrtf(vp[o] + EPSV);
    afv[o] = af;
    apv[o] = ap;
    cfv[o] = bf[o] * af + betaf[o] - mf[o] * af;
    cpv[o] = bp[o] * ap + betap[o] - mp[o] * ap;
    wd4[4*o+0] = Wd[3*o+0] * ad;
    wd4[4*o+1] = Wd[3*o+1] * ad;
    wd4[4*o+2] = Wd[3*o+2] * ad;
    wd4[4*o+3] = cd;                 // bias column, multiplies appended delta=1
  }
  __syncthreads();

  // ---- Stage BN-scaled weights as f16 into padded LDS.
  for (int i = tid; i < 128 * 128; i += 128) {
    const int o = i >> 7, c = i & 127;
    wf16[o * WSTRIDE + c] = (_Float16)(Wf[i] * afv[o]);
    wp16[o * WSTRIDE + c] = (_Float16)(Wp[i] * apv[o]);
  }
  __syncthreads();

  const int wave = tid >> 5;
  const int lane = tid & 31;
  const int col  = lane & 15;          // tile column / A row-in-tile
  const int half = lane >> 4;          // lane half selects K sub-range / M+8
  const int group = blockIdx.x * 4 + wave;   // 0..2499, exact (625*4 == 2*20000/16)
  const int b  = group / 1250;
  const int n0 = (group - b * 1250) * 16;
  const int n  = n0 + col;

  _Float16* fbuf = (_Float16*)(smem + OFF_WAVE + wave * FB_BYTES);
  _Float16* pbuf = (_Float16*)(smem + OFF_WAVE + 4 * FB_BYTES + wave * FB_BYTES);

  const float* featsB = feats + (size_t)b * CINV * NN;
  const float* ptsB   = pts + (size_t)b * 3 * NN;
  const int*   knnB   = knn + ((size_t)b * NN + n) * KK;

  const float px = ptsB[n];
  const float py = ptsB[NN + n];
  const float pz = ptsB[2 * NN + n];

  v8f acc[8];
  #pragma unroll
  for (int m = 0; m < 8; ++m) acc[m] = (v8f){};

  #pragma unroll 1
  for (int k = 0; k < KK; ++k) {
    const int j = knnB[k];

    // Prefetch a spread of next column's cache lines into WGP$/L2 while this
    // iteration's 72 WMMAs run (32 lanes x stride-4 channels).
    if (k + 1 < KK) {
      const int jn = knnB[k + 1];
      __builtin_prefetch(featsB + jn + (size_t)(lane * 4) * NN, 0, 3);
    }

    const float qx = ptsB[j]          - px;
    const float qy = ptsB[NN + j]     - py;
    const float qz = ptsB[2 * NN + j] - pz;

    // ---- Gather feats column j (64 channels per lane-half) into LDS as f16,
    //      column-major tile [col][ch], stride 136 halves. 8 channels per step:
    //      4x v_cvt_pk_rtz_f16_f32 + one 16-B ds_store.
    {
      const float* src = featsB + j;
      const int ch0 = half * 64;
      _Float16* dst = fbuf + col * WSTRIDE + ch0;
      #pragma unroll 4
      for (int ii = 0; ii < 8; ++ii) {
        const int ch = ch0 + 8 * ii;
        v2hf p0 = __builtin_amdgcn_cvt_pkrtz(src[(size_t)(ch + 0) * NN],
                                             src[(size_t)(ch + 1) * NN]);
        v2hf p1 = __builtin_amdgcn_cvt_pkrtz(src[(size_t)(ch + 2) * NN],
                                             src[(size_t)(ch + 3) * NN]);
        v2hf p2 = __builtin_amdgcn_cvt_pkrtz(src[(size_t)(ch + 4) * NN],
                                             src[(size_t)(ch + 5) * NN]);
        v2hf p3 = __builtin_amdgcn_cvt_pkrtz(src[(size_t)(ch + 6) * NN],
                                             src[(size_t)(ch + 7) * NN]);
        *(v8hf*)(dst + 8 * ii) = pack8(p0, p1, p2, p3);
      }
    }
    WAVE_LDS_FENCE();   // per-wave buffer; DS pipe is in-order within a wave

    // ---- Phase 1: feats GEMM, q-outer so the B tile is loaded once per chunk
    //      and shared across all 8 m-tiles.
    v8f accf[8];
    #pragma unroll
    for (int m = 0; m < 8; ++m) accf[m] = (v8f){};
    #pragma unroll
    for (int q = 0; q < 4; ++q) {
      // B (32x16 f16): lane col; 16 consecutive K per lane-half.
      const v16h Bv = ld_op(fbuf + col * WSTRIDE + q * 32 + half * 16, 8);
      #pragma unroll
      for (int m = 0; m < 8; ++m) {
        // A (16x32 f16): lane row; 16 consecutive K per lane-half.
        const v16h A = ld_op(wf16 + (m * 16 + col) * WSTRIDE + q * 32 + half * 16, 8);
        accf[m] = __builtin_amdgcn_wmma_f32_16x16x32_f16(false, A, false, Bv,
                                                         (short)0, accf[m], false, false);
      }
    }

    // ---- Delta conv (f32 WMMA, K=4 incl. bias column) + fused finalize.
    // f32 B 4x16: lanes0-15 VGPR{0,1}=K{0,1}; lanes16-31 VGPR{0,1}=K{2,3}.
    v2f bD;
    bD[0] = half ? qz : qx;
    bD[1] = half ? 1.0f : qy;
    #pragma unroll
    for (int m = 0; m < 8; ++m) {
      const int row = m * 16 + col;
      v2f aW;
      { const float* w = wd4 + row * 4 + half * 2; aW[0] = w[0]; aW[1] = w[1]; }
      v8f accd = (v8f){};
      accd = __builtin_amdgcn_wmma_f32_16x16x4_f32(false, aW, false, bD,
                                                   (short)0, accd, false, false);
      const v8f cf8 = *(const v8f*)(cfv + m * 16 + half * 8);
      float pr[8];
      #pragma unroll
      for (int v = 0; v < 8; ++v) {
        float dv = accd[v];             dv = dv > 0.f ? dv : 0.f;
        float fv = accf[m][v] + cf8[v]; fv = fv > 0.f ? fv : 0.f;
        pr[v] = dv * fv;
      }
      v2hf q0 = __builtin_amdgcn_cvt_pkrtz(pr[0], pr[1]);
      v2hf q1 = __builtin_amdgcn_cvt_pkrtz(pr[2], pr[3]);
      v2hf q2 = __builtin_amdgcn_cvt_pkrtz(pr[4], pr[5]);
      v2hf q3 = __builtin_amdgcn_cvt_pkrtz(pr[6], pr[7]);
      *(v8hf*)(pbuf + col * WSTRIDE + m * 16 + half * 8) = pack8(q0, q1, q2, q3);
    }
    WAVE_LDS_FENCE();

    // ---- Phase 2: post GEMM (Wp), q-outer again; accumulate K-sum in regs.
    v8f accp[8];
    #pragma unroll
    for (int m = 0; m < 8; ++m) accp[m] = (v8f){};
    #pragma unroll
    for (int q = 0; q < 4; ++q) {
      const v16h Bv = ld_op(pbuf + col * WSTRIDE + q * 32 + half * 16, 8);
      #pragma unroll
      for (int m = 0; m < 8; ++m) {
        const v16h A = ld_op(wp16 + (m * 16 + col) * WSTRIDE + q * 32 + half * 16, 8);
        accp[m] = __builtin_amdgcn_wmma_f32_16x16x32_f16(false, A, false, Bv,
                                                         (short)0, accp[m], false, false);
      }
    }
    #pragma unroll
    for (int m = 0; m < 8; ++m) {
      const v8f cp8 = *(const v8f*)(cpv + m * 16 + half * 8);
      #pragma unroll
      for (int v = 0; v < 8; ++v) {
        const float yv = accp[m][v] + cp8[v];
        acc[m][v] += (yv > 0.f ? yv : 0.f);
      }
    }
  }

  // ---- Epilogue: C layout -> out[b, o, n]; lanes 0-15 coalesce along n.
  float* outB = out + (size_t)b * GV * NN;
  #pragma unroll
  for (int m = 0; m < 8; ++m) {
    #pragma unroll
    for (int v = 0; v < 8; ++v) {
      const int o = m * 16 + half * 8 + v;
      outB[(size_t)o * NN + n] = acc[m][v];
    }
  }
}

extern "C" void kernel_launch(void* const* d_in, const int* in_sizes, int n_in,
                              void* d_out, int out_size, void* d_ws, size_t ws_size,
                              hipStream_t stream) {
  (void)in_sizes; (void)n_in; (void)out_size; (void)d_ws; (void)ws_size;
  const float* feats = (const float*)d_in[0];
  const float* pts   = (const float*)d_in[1];
  const int*   knn   = (const int*)  d_in[2];
  const float* Wd    = (const float*)d_in[3];
  const float* bd    = (const float*)d_in[4];
  const float* gd    = (const float*)d_in[5];
  const float* betad = (const float*)d_in[6];
  const float* md    = (const float*)d_in[7];
  const float* vd    = (const float*)d_in[8];
  const float* Wf    = (const float*)d_in[9];
  const float* bf    = (const float*)d_in[10];
  const float* gf    = (const float*)d_in[11];
  const float* betaf = (const float*)d_in[12];
  const float* mf    = (const float*)d_in[13];
  const float* vf    = (const float*)d_in[14];
  const float* Wp    = (const float*)d_in[15];
  const float* bp    = (const float*)d_in[16];
  const float* gp    = (const float*)d_in[17];
  const float* betap = (const float*)d_in[18];
  const float* mp    = (const float*)d_in[19];
  const float* vp    = (const float*)d_in[20];
  float* outp = (float*)d_out;

  (void)hipFuncSetAttribute((const void*)p3dconv_wmma,
                            hipFuncAttributeMaxDynamicSharedMemorySize, SMEM_TOTAL);

  // 2*20000/16 = 2500 point-groups, 4 waves (128 thr) per block -> 625 blocks.
  p3dconv_wmma<<<625, 128, SMEM_TOTAL, stream>>>(
      feats, pts, knn,
      Wd, bd, gd, betad, md, vd,
      Wf, bf, gf, betaf, mf, vf,
      Wp, bp, gp, betap, mp, vp,
      outp);
}
